// DinAttentionLayer_40209483825174
// MI455X (gfx1250) — matrix operand order repaired and checked by reading, not compiled, not run on hardware
//
#include <hip/hip_runtime.h>
#include <math.h>

typedef __bf16 v16bf __attribute__((ext_vector_type(16)));
typedef float  v8f   __attribute__((ext_vector_type(8)));
typedef int    i4v   __attribute__((ext_vector_type(4)));
typedef __attribute__((address_space(1))) i4v gi4;   // global int4
typedef __attribute__((address_space(3))) i4v li4;   // LDS int4

union Frag16 { v16bf v; unsigned short u[16]; uint4 q[2]; };

#if defined(__has_builtin)
# if __has_builtin(__builtin_amdgcn_global_load_async_to_lds_b128)
#  define HAVE_ASYNC_LDS 1
# endif
#endif
#ifndef HAVE_ASYNC_LDS
# define HAVE_ASYNC_LDS 0
#endif

static __device__ __forceinline__ unsigned int pkbf(float a, float b) {
  union { __bf16 h[2]; unsigned int u; } c;
  c.h[0] = (__bf16)a; c.h[1] = (__bf16)b;
  return c.u;
}
static __device__ __forceinline__ float sigmoidf_(float x) {
  return 1.0f / (1.0f + __expf(-x));
}

#define T_  200
#define E_  36

// LDS layout (bytes)
#define OFF_UB    0        // float[7200]          28800
#define OFF_W1T   28800    // bf16[80*160]         25600  (W1^T, K padded 144->160 with zeros)
#define OFF_W2T   54400    // bf16[48*96]           9216  (W2^T, K 80->96, N 40->48, zero pad)
#define OFF_Q     63616    // float[36]
#define OFF_B1    63760    // float[80]
#define OFF_B2    64080    // float[48]
#define OFF_W3    64272    // float[40]
#define OFF_B3    64432    // float[1] (pad to 16B)
#define OFF_SC    64448    // float[208] scores
#define OFF_RED   65280    // float[256] reduction
#define OFF_SCR   66304    // bf16[8*2560] per-wave 16x160 scratch (40960)
#define SMEM_BYTES 107264

extern "C" __global__ void __launch_bounds__(256, 1)
din_attention_kernel(const float* __restrict__ qad,
                     const float* __restrict__ ub,
                     const long long* __restrict__ lens,
                     const float* __restrict__ W1, const float* __restrict__ b1,
                     const float* __restrict__ W2, const float* __restrict__ b2,
                     const float* __restrict__ W3, const float* __restrict__ b3,
                     float* __restrict__ out)
{
  extern __shared__ char smem[];
  float*          s_ub  = (float*)(smem + OFF_UB);
  unsigned short* s_w1t = (unsigned short*)(smem + OFF_W1T);
  unsigned short* s_w2t = (unsigned short*)(smem + OFF_W2T);
  float*          s_q   = (float*)(smem + OFF_Q);
  float*          s_b1  = (float*)(smem + OFF_B1);
  float*          s_b2  = (float*)(smem + OFF_B2);
  float*          s_w3  = (float*)(smem + OFF_W3);
  float*          s_b3  = (float*)(smem + OFF_B3);
  float*          s_sc  = (float*)(smem + OFF_SC);
  float*          s_red = (float*)(smem + OFF_RED);
  unsigned short* s_scr = (unsigned short*)(smem + OFF_SCR);

  const int b   = blockIdx.x;
  const int tid = threadIdx.x;
  const float* ubb = ub + (size_t)b * (T_*E_);

  // ---- stage user_behavior row (28.8 KB) into LDS ----
#if HAVE_ASYNC_LDS
  for (int i = tid * 16; i < T_*E_*4; i += 256 * 16) {
    __builtin_amdgcn_global_load_async_to_lds_b128(
        (gi4*)((const char*)ubb + i),
        (li4*)(smem + OFF_UB + i), 0, 0);
  }
# if __has_builtin(__builtin_amdgcn_s_wait_asynccnt)
  __builtin_amdgcn_s_wait_asynccnt(0);
# else
  asm volatile("s_wait_asynccnt 0x0" ::: "memory");
# endif
#else
  for (int i = tid; i < T_*E_; i += 256) s_ub[i] = ubb[i];
#endif

  // ---- stage q + weights (bf16, transposed, zero-padded) ----
  for (int i = tid; i < E_; i += 256) s_q[i] = qad[(size_t)b*E_ + i];

  unsigned int* w1w = (unsigned int*)s_w1t;
  unsigned int* w2w = (unsigned int*)s_w2t;
  // zero-pad regions (disjoint from valid fills below; no barrier needed)
  for (int i = tid; i < 640; i += 256) w1w[(i >> 3)*80 + 72 + (i & 7)] = 0;   // k 144..159
  for (int i = tid; i < 320; i += 256) w2w[(i >> 3)*48 + 40 + (i & 7)] = 0;   // col<40, k 80..95
  for (int i = tid; i < 384; i += 256) w2w[(40 + i/48)*48 + (i % 48)] = 0;    // col 40..47

  __bf16* w1b = (__bf16*)s_w1t;
  for (int i = tid; i < 144*80; i += 256) {       // i = k*80 + col (coalesced read)
    int k = i / 80, col = i - k*80;
    w1b[col*160 + k] = (__bf16)W1[i];
  }
  __bf16* w2b = (__bf16*)s_w2t;
  for (int i = tid; i < 80*40; i += 256) {        // i = k*40 + col (coalesced read)
    int k = i / 40, col = i - k*40;
    w2b[col*96 + k] = (__bf16)W2[i];
  }
  if (tid < 80) s_b1[tid] = b1[tid];
  if (tid < 48) s_b2[tid] = (tid < 40) ? b2[tid] : 0.0f;
  if (tid < 40) s_w3[tid] = W3[tid];
  if (tid == 0) s_b3[0] = b3[0];
  __syncthreads();

  const int wave = tid >> 5;
  const int lane = tid & 31;
  const int half = lane >> 4;     // A frag: 0 -> K 0-7/16-23 ; 1 -> K 8-15/24-31
  const int lm   = lane & 15;
  unsigned short* scr  = s_scr + wave * 2560;     // 16 x 160 bf16
  unsigned int*   scrw = (unsigned int*)scr;
  __bf16*         scrb = (__bf16*)scr;

  for (int mt = wave; mt < 13; mt += 8) {         // 13 row tiles of 16 (208 >= 200)
    // ---- build x tile: [q | ub | q-ub | q*ub] as packed bf16 dwords ----
    for (int g = lane; g < 16*18; g += 32) {      // (row, e-pair)
      int r = g / 18, e = (g - r*18) << 1;
      int t = mt*16 + r;
      float q0 = 0.f, q1 = 0.f, u0 = 0.f, u1 = 0.f;
      if (t < T_) {
        float2 qv = *(const float2*)(s_q + e);
        float2 uv = *(const float2*)(s_ub + t*E_ + e);
        q0 = qv.x; q1 = qv.y; u0 = uv.x; u1 = uv.y;
      }
      int base = (r*160 + e) >> 1;
      scrw[base]      = pkbf(q0, q1);
      scrw[base + 18] = pkbf(u0, u1);
      scrw[base + 36] = pkbf(q0 - u0, q1 - u1);
      scrw[base + 54] = pkbf(q0 * u0, q1 * u1);
    }
    for (int i = lane; i < 16*8; i += 32)          // pad cols 144..159
      scrw[(i >> 3)*80 + 72 + (i & 7)] = 0;

    // ---- layer 1: [16x160]x[160x80], 5 Ntiles x 5 Ksteps ----
    v8f acc[5] = {};
    #pragma unroll
    for (int ks = 0; ks < 5; ++ks) {
      const int kb = ks * 32;
      Frag16 a;
      const unsigned short* ar = scr + lm*160 + kb + half*8;
      a.q[0] = *(const uint4*)(ar);
      a.q[1] = *(const uint4*)(ar + 16);
      #pragma unroll
      for (int nt = 0; nt < 5; ++nt) {
        Frag16 w;
        const unsigned short* br = s_w1t + (nt*16 + lm)*160 + kb + half*8;
        w.q[0] = *(const uint4*)(br);
        w.q[1] = *(const uint4*)(br + 16);
        acc[nt] = __builtin_amdgcn_wmma_f32_16x16x32_bf16(false, a.v, false, w.v,
                                                          (short)0, acc[nt], false, false);
      }
    }
    // sigmoid(+b1) -> h1 as [16 x 96] bf16 (cols 80..95 zero)
    #pragma unroll
    for (int nt = 0; nt < 5; ++nt) {
      const int n = nt*16 + lm;
      const float bias = s_b1[n];
      #pragma unroll
      for (int r = 0; r < 8; ++r)
        scrb[(r + half*8)*96 + n] = (__bf16)sigmoidf_(acc[nt][r] + bias);
    }
    for (int i = lane; i < 16*8; i += 32)
      scrw[(i >> 3)*48 + 40 + (i & 7)] = 0;

    // ---- layer 2: [16x96]x[96x48], 3 Ntiles x 3 Ksteps ----
    v8f acc2[3] = {};
    #pragma unroll
    for (int ks = 0; ks < 3; ++ks) {
      const int kb = ks * 32;
      Frag16 a;
      const unsigned short* ar = scr + lm*96 + kb + half*8;
      a.q[0] = *(const uint4*)(ar);
      a.q[1] = *(const uint4*)(ar + 16);
      #pragma unroll
      for (int nt = 0; nt < 3; ++nt) {
        Frag16 w;
        const unsigned short* br = s_w2t + (nt*16 + lm)*96 + kb + half*8;
        w.q[0] = *(const uint4*)(br);
        w.q[1] = *(const uint4*)(br + 16);
        acc2[nt] = __builtin_amdgcn_wmma_f32_16x16x32_bf16(false, a.v, false, w.v,
                                                           (short)0, acc2[nt], false, false);
      }
    }
    // sigmoid(+b2) -> h2 at bf16[1536 + 16x48]
    #pragma unroll
    for (int nt = 0; nt < 3; ++nt) {
      const int n = nt*16 + lm;
      const float bias = s_b2[n];
      #pragma unroll
      for (int r = 0; r < 8; ++r)
        scrb[1536 + (r + half*8)*48 + n] = (__bf16)sigmoidf_(acc2[nt][r] + bias);
    }

    // ---- layer 3: 40-wide dot per row ----
    if (lane < 16) {
      const int t = mt*16 + lane;
      if (t < T_) {
        float sc = s_b3[0];
        const __bf16* hr = scrb + 1536 + lane*48;
        #pragma unroll
        for (int o = 0; o < 40; ++o) sc += (float)hr[o] * s_w3[o];
        s_sc[t] = sc;
      }
    }
  }
  __syncthreads();

  // ---- multiplicative mask + softmax over all T (as in reference) ----
  const long long len = lens[b];
  float sval = 0.0f;
  if (tid < T_) sval = (tid < (int)len) ? s_sc[tid] : 0.0f;
  s_red[tid] = (tid < T_) ? sval : -3.4e38f;
  __syncthreads();
  for (int off = 128; off > 0; off >>= 1) {
    if (tid < off) s_red[tid] = fmaxf(s_red[tid], s_red[tid + off]);
    __syncthreads();
  }
  const float mx = s_red[0];
  __syncthreads();
  const float p = (tid < T_) ? __expf(sval - mx) : 0.0f;
  s_red[tid] = p;
  __syncthreads();
  for (int off = 128; off > 0; off >>= 1) {
    if (tid < off) s_red[tid] += s_red[tid + off];
    __syncthreads();
  }
  const float denom = s_red[0];
  __syncthreads();
  if (tid < T_) s_sc[tid] = p / denom;
  __syncthreads();

  // ---- out[b, e] = sum_t w_t * ub[b, t, e] ----
  if (tid < E_) {
    float acc = 0.0f;
    for (int t = 0; t < T_; ++t) acc += s_sc[t] * s_ub[t*E_ + tid];
    out[(size_t)b*E_ + tid] = acc;
  }
}

extern "C" void kernel_launch(void* const* d_in, const int* in_sizes, int n_in,
                              void* d_out, int out_size, void* d_ws, size_t ws_size,
                              hipStream_t stream) {
  (void)in_sizes; (void)n_in; (void)d_ws; (void)ws_size; (void)out_size;
  const float*     qad  = (const float*)d_in[0];
  const float*     ubp  = (const float*)d_in[1];
  const long long* lens = (const long long*)d_in[2];
  const float*     W1   = (const float*)d_in[3];
  const float*     b1   = (const float*)d_in[4];
  const float*     W2   = (const float*)d_in[5];
  const float*     b2   = (const float*)d_in[6];
  const float*     W3   = (const float*)d_in[7];
  const float*     b3   = (const float*)d_in[8];
  float* out = (float*)d_out;

  (void)hipFuncSetAttribute((const void*)din_attention_kernel,
                            hipFuncAttributeMaxDynamicSharedMemorySize, SMEM_BYTES);

  hipLaunchKernelGGL(din_attention_kernel, dim3(4096), dim3(256), SMEM_BYTES, stream,
                     qad, ubp, lens, W1, b1, W2, b2, W3, b3, out);
}
